// Qwen35MoEBlock_22797686407845
// MI455X (gfx1250) — matrix operand order repaired and checked by reading, not compiled, not run on hardware
//
#include <hip/hip_runtime.h>
#include <math.h>

// ---------------- problem constants (match reference) ----------------
#define Bq   2
#define Sq   1024
#define Tq   (Bq*Sq)      // 2048 tokens
#define Hq   2048
#define Eq   8
#define Iq   1408
#define ISq  5632

// ---------------- WMMA types ----------------
typedef __bf16 v16bf __attribute__((ext_vector_type(16)));
typedef float  v8f   __attribute__((ext_vector_type(8)));

union Frag {
    v16bf v;
    uint4 q[2];
};

// GEMM tiling: block 128M x 64N x 32K, 256 threads = 8 waves,
// wave grid 4(M) x 2(N), each wave owns 32x32 = 2x2 WMMA(16x16x32) tiles.
#define BM 128
#define BN 64
#define BK 32
#define LDA 40   // bf16 elems/row: 64B data + 16B pad = 80B (16B aligned rows)
#define LDB 40

// ---------------- CDNA5 TDM support probes ----------------
#if defined(__has_builtin)
#  if __has_builtin(__builtin_amdgcn_tensor_load_to_lds)
#    define HAVE_TDM 1
#  endif
#  if __has_builtin(__builtin_amdgcn_s_wait_tensorcnt)
#    define HAVE_TCNT 1
#  endif
#endif

typedef unsigned int u32x4 __attribute__((ext_vector_type(4)));
typedef int          i32x4 __attribute__((ext_vector_type(4)));
typedef int          i32x8 __attribute__((ext_vector_type(8)));

__device__ __forceinline__ void wait_tensor0() {
#ifdef HAVE_TCNT
    __builtin_amdgcn_s_wait_tensorcnt(0);
#else
    asm volatile("s_wait_tensorcnt 0x0" ::: "memory");
#endif
}

#ifdef HAVE_TDM
// Issue one TDM descriptor: load a rows x 32 (bf16) tile from global into LDS,
// LDS rows padded 64B data + 16B pad (pad_interval=3 -> every 64B, pad_amount=3 -> 16B)
// so the LDS layout matches LDA=40 bf16 elements per row.
__device__ __forceinline__ void tdm_load_a_tile(unsigned lds_off, const void* gaddr,
                                                unsigned rows, unsigned stride_elems,
                                                unsigned tensor_d0, unsigned tensor_d1) {
    unsigned long long ga = (unsigned long long)(uintptr_t)gaddr;
    u32x4 g0;
    g0[0] = 1u;                                               // count=1, user mode
    g0[1] = lds_off;                                          // lds_addr (bytes)
    g0[2] = (unsigned)(ga & 0xFFFFFFFFu);                     // global_addr[31:0]
    g0[3] = (unsigned)((ga >> 32) & 0x01FFFFFFu) | (2u << 30);// global_addr[56:32] | type=2
    i32x8 g1;
    g1[0] = (int)((1u << 16) | (1u << 20) | (3u << 22) | (3u << 25));
    //        data_size=2B     pad_enable   pad_int=64B  pad_amt=16B ; wg_mask=0
    g1[1] = (int)((tensor_d0 & 0xFFFFu) << 16);               // tensor_dim0[15:0]
    g1[2] = (int)((tensor_d0 >> 16) | ((tensor_d1 & 0xFFFFu) << 16));
    g1[3] = (int)((tensor_d1 >> 16) | (32u << 16));           // tile_dim0 = 32 elems
    g1[4] = (int)rows;                                        // tile_dim1 = rows, tile_dim2=0
    g1[5] = (int)stride_elems;                                // tensor_dim0_stride[31:0]
    g1[6] = 0;                                                // stride[47:32] | dim1_stride lo
    g1[7] = 0;
    i32x4 gz = (i32x4)0;
#if defined(__clang_major__) && __clang_major__ >= 23
    i32x8 gz8 = (i32x8)0;
    __builtin_amdgcn_tensor_load_to_lds(g0, g1, gz, gz, gz8, 0);
#else
    __builtin_amdgcn_tensor_load_to_lds(g0, g1, gz, gz, 0);
#endif
}
#endif // HAVE_TDM

// ============================================================
// 1) fp32 -> bf16 convert (x -> xb)
// ============================================================
__global__ __launch_bounds__(256) void cvt_kernel(const float* __restrict__ x,
                                                  __bf16* __restrict__ xb) {
    int i = (blockIdx.x * 256 + threadIdx.x) * 8;
    float4 a = *(const float4*)(x + i);
    float4 b = *(const float4*)(x + i + 4);
    union { __bf16 h[8]; uint4 q; } u;
    u.h[0] = (__bf16)a.x; u.h[1] = (__bf16)a.y; u.h[2] = (__bf16)a.z; u.h[3] = (__bf16)a.w;
    u.h[4] = (__bf16)b.x; u.h[5] = (__bf16)b.y; u.h[6] = (__bf16)b.z; u.h[7] = (__bf16)b.w;
    *(uint4*)(xb + i) = u.q;
}

// ============================================================
// 2) router: softmax -> top-2 -> renormalized dense combine[T,E],
//    plus sigmoid shared-expert gate sgate[T].
// ============================================================
__global__ __launch_bounds__(256) void router_kernel(const float* __restrict__ x,
                                                     const float* __restrict__ gw,
                                                     const float* __restrict__ sgw,
                                                     float* __restrict__ combine,
                                                     float* __restrict__ sgate) {
    const int t    = blockIdx.x;
    const int tid  = threadIdx.x;
    const int wave = tid >> 5;
    const int lane = tid & 31;
    const float* xr = x + (size_t)t * Hq;

    __shared__ float slog[Eq];
    __shared__ float sgacc;
    if (tid == 0) sgacc = 0.f;
    __syncthreads();

    float p = 0.f;
    const float* gwr = gw + (size_t)wave * Hq;
    for (int h = lane; h < Hq; h += 32) p += xr[h] * gwr[h];
    #pragma unroll
    for (int off = 16; off > 0; off >>= 1) p += __shfl_xor(p, off, 32);
    if (lane == 0) slog[wave] = p;

    float q = 0.f;
    for (int h = tid; h < Hq; h += 256) q += xr[h] * sgw[h];
    #pragma unroll
    for (int off = 16; off > 0; off >>= 1) q += __shfl_xor(q, off, 32);
    if (lane == 0) atomicAdd(&sgacc, q);
    __syncthreads();

    if (tid == 0) {
        float l[Eq], pr[Eq];
        float m = -1e30f;
        #pragma unroll
        for (int e = 0; e < Eq; ++e) { l[e] = slog[e]; m = fmaxf(m, l[e]); }
        float s = 0.f;
        #pragma unroll
        for (int e = 0; e < Eq; ++e) { pr[e] = __expf(l[e] - m); s += pr[e]; }
        float inv = 1.f / s;
        #pragma unroll
        for (int e = 0; e < Eq; ++e) pr[e] *= inv;
        int a = 0;
        #pragma unroll
        for (int e = 1; e < Eq; ++e) if (pr[e] > pr[a]) a = e;
        int b = (a == 0) ? 1 : 0;
        #pragma unroll
        for (int e = 0; e < Eq; ++e) if (e != a && pr[e] > pr[b]) b = e;
        float rn = 1.f / (pr[a] + pr[b]);
        #pragma unroll
        for (int e = 0; e < Eq; ++e)
            combine[(size_t)t * Eq + e] = (e == a) ? pr[a] * rn : ((e == b) ? pr[b] * rn : 0.f);
        sgate[t] = 1.f / (1.f + __expf(-sgacc));
    }
}

// ============================================================
// 3) fused gate+up WMMA GEMM:  Out[t,n] = bf16( silu(A@Wg) * (A@Wu) * scale )
//    SS==Eq : expert path, scale = combine[row*E + z]
//    SS==1  : shared path, scale = sgate[row]
//    A tile staged via TDM (tensor_load_to_lds) when available.
// ============================================================
template <int SS>
__global__ __launch_bounds__(256) void gateup_kernel(const __bf16* __restrict__ A,
                                                     const float* __restrict__ Wg,
                                                     const float* __restrict__ Wu,
                                                     __bf16* __restrict__ Out,
                                                     const float* __restrict__ scale,
                                                     int N, int Kdim) {
    const int z = blockIdx.z;
    Wg  += (size_t)z * Kdim * N;
    Wu  += (size_t)z * Kdim * N;
    Out += (size_t)z * Tq * N;
    const int gm0 = blockIdx.x * BM;
    const int gn0 = blockIdx.y * BN;

    __shared__ __bf16 sA [BM * LDA];
    __shared__ __bf16 sBg[BN * LDB];
    __shared__ __bf16 sBu[BN * LDB];

    const int tid  = threadIdx.x;
    const int wave = tid >> 5, lane = tid & 31;
    const int wm = wave >> 1, wn = wave & 1;
    const int half = lane >> 4, l16 = lane & 15;

    const int bk = tid >> 3,  bn = (tid & 7) * 8;       // B: 8 threads/row, 8 fp32 each
#ifndef HAVE_TDM
    const int arow = tid >> 1, aseg = tid & 1;          // A fallback: 2 threads/row, 32B each
#endif
#ifdef HAVE_TDM
    const unsigned sA_off = (unsigned)(uintptr_t)(void*)&sA[0];
#endif

    v8f accG[2][2] = {}, accU[2][2] = {};

    for (int k0 = 0; k0 < Kdim; k0 += BK) {
        // ---- stage A tile: TDM async DMA (bf16 rows, hardware pad -> LDA) ----
#ifdef HAVE_TDM
        if (wave == 0)
            tdm_load_a_tile(sA_off, A + (size_t)gm0 * Kdim + k0, BM,
                            (unsigned)Kdim, (unsigned)Kdim, (unsigned)Tq);
#else
        const uint4* ap = (const uint4*)(A + (size_t)(gm0 + arow) * Kdim + k0 + aseg * 16);
        *(uint4*)&sA[arow * LDA + aseg * 16]     = ap[0];
        *(uint4*)&sA[arow * LDA + aseg * 16 + 8] = ap[1];
#endif
        // ---- stage Wg/Wu tiles: fp32 -> bf16, transposed to [N][K] ----
        const float4* wgp = (const float4*)(Wg + (size_t)(k0 + bk) * N + gn0 + bn);
        const float4* wup = (const float4*)(Wu + (size_t)(k0 + bk) * N + gn0 + bn);
        float4 g0 = wgp[0], g1 = wgp[1], u0 = wup[0], u1 = wup[1];
        if (k0 + BK < Kdim) {
            __builtin_prefetch(Wg + (size_t)(k0 + BK + bk) * N + gn0 + bn, 0, 1);
            __builtin_prefetch(Wu + (size_t)(k0 + BK + bk) * N + gn0 + bn, 0, 1);
        }
        float gv[8] = {g0.x, g0.y, g0.z, g0.w, g1.x, g1.y, g1.z, g1.w};
        float uv[8] = {u0.x, u0.y, u0.z, u0.w, u1.x, u1.y, u1.z, u1.w};
        #pragma unroll
        for (int j = 0; j < 8; ++j) {
            sBg[(bn + j) * LDB + bk] = (__bf16)gv[j];
            sBu[(bn + j) * LDB + bk] = (__bf16)uv[j];
        }
#ifdef HAVE_TDM
        if (wave == 0) wait_tensor0();
#endif
        __syncthreads();

        // ---- fragments ----
        Frag af[2], bg[2], bu[2];
        #pragma unroll
        for (int tm = 0; tm < 2; ++tm) {
            int r = wm * 32 + tm * 16 + l16;
            af[tm].q[0] = *(const uint4*)&sA[r * LDA + half * 8];       // K 0..7 / 8..15
            af[tm].q[1] = *(const uint4*)&sA[r * LDA + 16 + half * 8];  // K 16..23 / 24..31
        }
        #pragma unroll
        for (int tn = 0; tn < 2; ++tn) {
            int n = wn * 32 + tn * 16 + l16;
            bg[tn].q[0] = *(const uint4*)&sBg[n * LDB + half * 8];
            bg[tn].q[1] = *(const uint4*)&sBg[n * LDB + 16 + half * 8];
            bu[tn].q[0] = *(const uint4*)&sBu[n * LDB + half * 8];
            bu[tn].q[1] = *(const uint4*)&sBu[n * LDB + 16 + half * 8];
        }
        #pragma unroll
        for (int tm = 0; tm < 2; ++tm)
            #pragma unroll
            for (int tn = 0; tn < 2; ++tn) {
                accG[tm][tn] = __builtin_amdgcn_wmma_f32_16x16x32_bf16(
                    false, af[tm].v, false, bg[tn].v, (short)0, accG[tm][tn], false, false);
                accU[tm][tn] = __builtin_amdgcn_wmma_f32_16x16x32_bf16(
                    false, af[tm].v, false, bu[tn].v, (short)0, accU[tm][tn], false, false);
            }
        __syncthreads();
    }

    // ---- epilogue: silu(g)*u*scale -> bf16 ----
    #pragma unroll
    for (int tm = 0; tm < 2; ++tm)
        #pragma unroll
        for (int tn = 0; tn < 2; ++tn)
            #pragma unroll
            for (int r = 0; r < 8; ++r) {
                int gm = gm0 + wm * 32 + tm * 16 + half * 8 + r;
                int gn = gn0 + wn * 32 + tn * 16 + l16;
                float g = accG[tm][tn][r];
                float u = accU[tm][tn][r];
                float s = (SS == Eq) ? scale[(size_t)gm * Eq + z] : scale[gm];
                float h = (g / (1.f + __expf(-g))) * u * s;
                Out[(size_t)gm * N + gn] = (__bf16)h;
            }
}

// ============================================================
// 4) down-projection WMMA GEMM over concatenated experts:
//    Out[t,h] (=|+=) sum_k A[e(k), t, i(k)] * W[k, h]
// ============================================================
template <bool ACC>
__global__ __launch_bounds__(256) void down_kernel(const __bf16* __restrict__ A,
                                                   const float* __restrict__ W,
                                                   float* __restrict__ Out,
                                                   int Ktot, int Iinner) {
    const int N = Hq;
    const int gm0 = blockIdx.x * BM;
    const int gn0 = blockIdx.y * BN;

    __shared__ __bf16 sA[BM * LDA];
    __shared__ __bf16 sB[BN * LDB];

    const int tid  = threadIdx.x;
    const int wave = tid >> 5, lane = tid & 31;
    const int wm = wave >> 1, wn = wave & 1;
    const int half = lane >> 4, l16 = lane & 15;
    const int bk = tid >> 3,  bn = (tid & 7) * 8;
#ifndef HAVE_TDM
    const int arow = tid >> 1, aseg = tid & 1;
#endif
#ifdef HAVE_TDM
    const unsigned sA_off = (unsigned)(uintptr_t)(void*)&sA[0];
#endif

    v8f acc[2][2] = {};

    for (int k0 = 0; k0 < Ktot; k0 += BK) {
        int e  = k0 / Iinner;            // Iinner % 32 == 0 -> K-step stays in one expert
        int i0 = k0 - e * Iinner;
#ifdef HAVE_TDM
        if (wave == 0)
            tdm_load_a_tile(sA_off, A + ((size_t)e * Tq + gm0) * Iinner + i0, BM,
                            (unsigned)Iinner, (unsigned)Iinner, (unsigned)(Tq * Eq));
#else
        const uint4* ap = (const uint4*)(A + ((size_t)e * Tq + gm0 + arow) * Iinner + i0 + aseg * 16);
        *(uint4*)&sA[arow * LDA + aseg * 16]     = ap[0];
        *(uint4*)&sA[arow * LDA + aseg * 16 + 8] = ap[1];
#endif
        const float4* wp = (const float4*)(W + (size_t)(k0 + bk) * N + gn0 + bn);
        float4 w0 = wp[0], w1 = wp[1];
        if (k0 + BK < Ktot)
            __builtin_prefetch(W + (size_t)(k0 + BK + bk) * N + gn0 + bn, 0, 1);
        float wv[8] = {w0.x, w0.y, w0.z, w0.w, w1.x, w1.y, w1.z, w1.w};
        #pragma unroll
        for (int j = 0; j < 8; ++j) sB[(bn + j) * LDB + bk] = (__bf16)wv[j];
#ifdef HAVE_TDM
        if (wave == 0) wait_tensor0();
#endif
        __syncthreads();

        Frag af[2], bf[2];
        #pragma unroll
        for (int tm = 0; tm < 2; ++tm) {
            int r = wm * 32 + tm * 16 + l16;
            af[tm].q[0] = *(const uint4*)&sA[r * LDA + half * 8];
            af[tm].q[1] = *(const uint4*)&sA[r * LDA + 16 + half * 8];
        }
        #pragma unroll
        for (int tn = 0; tn < 2; ++tn) {
            int n = wn * 32 + tn * 16 + l16;
            bf[tn].q[0] = *(const uint4*)&sB[n * LDB + half * 8];
            bf[tn].q[1] = *(const uint4*)&sB[n * LDB + 16 + half * 8];
        }
        #pragma unroll
        for (int tm = 0; tm < 2; ++tm)
            #pragma unroll
            for (int tn = 0; tn < 2; ++tn)
                acc[tm][tn] = __builtin_amdgcn_wmma_f32_16x16x32_bf16(
                    false, af[tm].v, false, bf[tn].v, (short)0, acc[tm][tn], false, false);
        __syncthreads();
    }

    #pragma unroll
    for (int tm = 0; tm < 2; ++tm)
        #pragma unroll
        for (int tn = 0; tn < 2; ++tn)
            #pragma unroll
            for (int r = 0; r < 8; ++r) {
                int gm = gm0 + wm * 32 + tm * 16 + half * 8 + r;
                int gn = gn0 + wn * 32 + tn * 16 + l16;
                size_t idx = (size_t)gm * N + gn;
                float v = acc[tm][tn][r];
                if (ACC) Out[idx] += v; else Out[idx] = v;
            }
}

// ============================================================
// launch
// ============================================================
extern "C" void kernel_launch(void* const* d_in, const int* in_sizes, int n_in,
                              void* d_out, int out_size, void* d_ws, size_t ws_size,
                              hipStream_t stream) {
    const float* x    = (const float*)d_in[0];   // [B,S,H]
    const float* gw   = (const float*)d_in[1];   // [E,H]
    const float* wg   = (const float*)d_in[2];   // [E,H,I]
    const float* wu   = (const float*)d_in[3];   // [E,H,I]
    const float* wd   = (const float*)d_in[4];   // [E,I,H]
    const float* swg  = (const float*)d_in[5];   // [H,IS]
    const float* swu  = (const float*)d_in[6];   // [H,IS]
    const float* swd  = (const float*)d_in[7];   // [IS,H]
    const float* sgw  = (const float*)d_in[8];   // [H,1]
    float* out = (float*)d_out;                  // [B,S,H] fp32

    // workspace layout (bytes)
    char* ws = (char*)d_ws;
    __bf16* xb      = (__bf16*)(ws);                                  //  8,388,608
    float*  combine = (float*)(ws + 8388608);                         //     65,536
    float*  sgate   = (float*)(ws + 8388608 + 65536);                 //      8,192
    __bf16* hb      = (__bf16*)(ws + 8462336);                        // 46,137,344  [E,T,I]
    __bf16* shb     = (__bf16*)(ws + 8462336 + 46137344);             // 23,068,672  [T,IS]

    cvt_kernel<<<(Tq * Hq) / (256 * 8), 256, 0, stream>>>(x, xb);
    router_kernel<<<Tq, 256, 0, stream>>>(x, gw, sgw, combine, sgate);
    gateup_kernel<Eq><<<dim3(Tq / BM, Iq / BN, Eq), 256, 0, stream>>>(
        xb, wg, wu, hb, combine, Iq, Hq);
    gateup_kernel<1><<<dim3(Tq / BM, ISq / BN, 1), 256, 0, stream>>>(
        xb, swg, swu, shb, sgate, ISq, Hq);
    down_kernel<false><<<dim3(Tq / BM, Hq / BN), 256, 0, stream>>>(
        hb, wd, out, Eq * Iq, Iq);
    down_kernel<true><<<dim3(Tq / BM, Hq / BN), 256, 0, stream>>>(
        shb, swd, out, ISq, ISq);
}
